// AnticipationDecoder_75814762709797
// MI455X (gfx1250) — matrix operand order repaired and checked by reading, not compiled, not run on hardware
//
#include <hip/hip_runtime.h>
#include <hip/hip_bf16.h>
#include <math.h>

#define Hh   512
#define NHh  8
#define DHh  64
#define FFh  2048
#define NOBJ 20
#define Pp   380
#define Bb   16
#define HPAD 520   // halves; 520*2=1040 bytes = 65*16 -> 16B aligned rows, conflict-spread
#define TM   32    // rows per workgroup in the fused pair kernel

typedef _Float16 h16v __attribute__((ext_vector_type(16)));
typedef _Float16 h8v  __attribute__((ext_vector_type(8)));
typedef float    f8v  __attribute__((ext_vector_type(8)));

__device__ __forceinline__ float gelu_exact(float x) {
    return 0.5f * x * (1.0f + erff(x * 0.70710678118654752f));
}

// ---------------- x init: broadcast future_queries ----------------
__global__ void init_x_kernel(const float* fq, float* x, int F) {
    int idx = blockIdx.x * blockDim.x + threadIdx.x;
    int total = Bb * F * Hh;
    if (idx >= total) return;
    int c = idx % Hh;
    int f = (idx / Hh) % F;
    x[idx] = fq[f * Hh + c];
}

// ---------------- generic small GEMM: Y = act(X @ W^T + bias) ----------------
// X: (M, ldx) uses first K cols.  W row n at W + n*ldw + woff (K floats).
__global__ void gemm_bias_act(const float* __restrict__ X, int ldx,
                              const float* __restrict__ W, int ldw, int woff,
                              const float* __restrict__ bias,
                              float* __restrict__ Y, int ldy,
                              int M, int N, int K, int act) {
    int idx = blockIdx.x * blockDim.x + threadIdx.x;
    if (idx >= M * N) return;
    int m = idx / N, n = idx % N;
    const float* xr = X + (size_t)m * ldx;
    const float* wr = W + (size_t)n * ldw + woff;
    float acc = 0.f;
    #pragma unroll 4
    for (int k = 0; k < K; k += 4) {
        acc += xr[k] * wr[k] + xr[k+1] * wr[k+1] + xr[k+2] * wr[k+2] + xr[k+3] * wr[k+3];
    }
    if (bias) acc += bias[n];
    if (act)  acc = gelu_exact(acc);
    Y[(size_t)m * ldy + n] = acc;
}

// ---------------- attention: one thread per (b, head, q) ----------------
__global__ void attn_kernel(const float* __restrict__ Qb, int ldq, int qoff,
                            const float* __restrict__ KVb, int ldkv, int koff, int voff,
                            float* __restrict__ Out,
                            int Tq, int Tk) {
    int idx = blockIdx.x * blockDim.x + threadIdx.x;
    if (idx >= Bb * NHh * Tq) return;
    int q = idx % Tq;
    int h = (idx / Tq) % NHh;
    int b = idx / (Tq * NHh);
    const float* qv = Qb + (size_t)(b * Tq + q) * ldq + qoff + h * DHh;
    float s[16];
    float mx = -1e30f;
    for (int t = 0; t < Tk; ++t) {
        const float* kv = KVb + (size_t)(b * Tk + t) * ldkv + koff + h * DHh;
        float d = 0.f;
        #pragma unroll
        for (int e = 0; e < DHh; ++e) d += qv[e] * kv[e];
        d *= 0.125f;  // 1/sqrt(64)
        s[t] = d;
        mx = fmaxf(mx, d);
    }
    float se = 0.f;
    for (int t = 0; t < Tk; ++t) { s[t] = expf(s[t] - mx); se += s[t]; }
    float inv = 1.f / se;
    float o[DHh];
    #pragma unroll
    for (int e = 0; e < DHh; ++e) o[e] = 0.f;
    for (int t = 0; t < Tk; ++t) {
        float wgt = s[t] * inv;
        const float* vv = KVb + (size_t)(b * Tk + t) * ldkv + voff + h * DHh;
        #pragma unroll
        for (int e = 0; e < DHh; ++e) o[e] += wgt * vv[e];
    }
    float* op = Out + (size_t)(b * Tq + q) * Hh + h * DHh;
    #pragma unroll
    for (int e = 0; e < DHh; ++e) op[e] = o[e];
}

// ---------------- LayerNorm(X [+ R]) * w + b, one block per row ----------------
__global__ __launch_bounds__(256)
void ln_kernel(const float* __restrict__ X, const float* __restrict__ R,
               const float* __restrict__ w, const float* __restrict__ bp,
               float* __restrict__ Y, int rows) {
    int row = blockIdx.x;
    if (row >= rows) return;
    int t = threadIdx.x;
    __shared__ float red[256];
    const float* xr = X + (size_t)row * Hh;
    const float* rr = R ? R + (size_t)row * Hh : nullptr;
    float v0 = xr[t]       + (rr ? rr[t]       : 0.f);
    float v1 = xr[t + 256] + (rr ? rr[t + 256] : 0.f);
    red[t] = v0 + v1;
    __syncthreads();
    for (int s2 = 128; s2 > 0; s2 >>= 1) { if (t < s2) red[t] += red[t + s2]; __syncthreads(); }
    float mean = red[0] * (1.f / Hh);
    __syncthreads();
    float d0 = v0 - mean, d1 = v1 - mean;
    red[t] = d0 * d0 + d1 * d1;
    __syncthreads();
    for (int s2 = 128; s2 > 0; s2 >>= 1) { if (t < s2) red[t] += red[t + s2]; __syncthreads(); }
    float inv = rsqrtf(red[0] * (1.f / Hh) + 1e-5f);
    Y[(size_t)row * Hh + t]       = d0 * inv * w[t]       + bp[t];
    Y[(size_t)row * Hh + t + 256] = d1 * inv * w[t + 256] + bp[t + 256];
}

// ---------------- f32 -> f16 convert ----------------
__global__ void cvt_f16_kernel(const float* src, _Float16* dst, int n) {
    int i = blockIdx.x * blockDim.x + threadIdx.x;
    if (i < n) dst[i] = (_Float16)src[i];
}

// ---------------- pack head weights: rows 0-25 pred_w, 26 ex_w, 27-31 zero ----------------
__global__ void build_head_kernel(const float* pred_w, const float* ex_w, _Float16* pw) {
    int i = blockIdx.x * blockDim.x + threadIdx.x;
    if (i >= 32 * Hh) return;
    int r = i / Hh, c = i % Hh;
    float v = (r < 26) ? pred_w[r * Hh + c] : (r == 26 ? ex_w[c] : 0.f);
    pw[i] = (_Float16)v;
}

// ---------------- fused pair kernel: gelu -> enc (WMMA) -> heads (WMMA) ----------------
// TM=32 rows per block; 256 threads = 8 wave32. Each wave: 32x64 slice of enc
// (2 M-tiles x 4 N-tiles), so every B-fragment load feeds two v_wmma ops and
// pe2 L2 traffic is halved vs 16-row tiles. hbuf is reused for enc (extra
// barrier) to stay within the 64KB static LDS budget.
__global__ __launch_bounds__(256)
void pair_head_kernel(const float* __restrict__ Sb, const float* __restrict__ Ob,
                      const float* __restrict__ Cb,
                      const float* __restrict__ pe2_b,
                      const _Float16* __restrict__ pe2h,
                      const _Float16* __restrict__ pwh,
                      const float* __restrict__ pred_b, const float* __restrict__ ex_b,
                      float* __restrict__ out_pred, float* __restrict__ out_ex,
                      int F, int R) {
    __shared__ _Float16 hbuf[TM * HPAD];
    int t  = threadIdx.x;
    int r0 = blockIdx.x * TM;

    // ---- phase 1: h = gelu(S[b,i] + O[b,j] + C[b,f]) (pe1_b folded into C) ----
    {
        int mr = t & 31;       // row in tile
        int cg = t >> 5;       // 8 column groups of 64
        int r  = r0 + mr;
        int cbase = cg * 64;
        if (r < R) {
            int p  = r % Pp;
            int bf = r / Pp;
            int f  = bf % F;
            int b  = bf / F;
            int i  = p / (NOBJ - 1);
            int jr = p % (NOBJ - 1);
            int j  = jr + (jr >= i ? 1 : 0);
            const float* sp = Sb + (size_t)(b * NOBJ + i) * Hh;
            const float* op = Ob + (size_t)(b * NOBJ + j) * Hh;
            const float* cp = Cb + (size_t)(b * F + f) * Hh;
            #pragma unroll 4
            for (int c = cbase; c < cbase + 64; ++c) {
                float v = sp[c] + op[c] + cp[c];
                hbuf[mr * HPAD + c] = (_Float16)gelu_exact(v);
            }
        } else {
            for (int c = cbase; c < cbase + 64; ++c) hbuf[mr * HPAD + c] = (_Float16)0.f;
        }
    }
    __syncthreads();

    int w    = t >> 5;       // wave 0..7
    int lane = t & 31;
    int ml   = lane & 15;
    int sel  = lane >> 4;    // lane-half select

    // ---- phase 2: enc = h @ pe2^T  (each wave: 32 x 64 slice) ----
    f8v acc[2][4];
    #pragma unroll
    for (int mt = 0; mt < 2; ++mt)
        #pragma unroll
        for (int nt = 0; nt < 4; ++nt)
            acc[mt][nt] = (f8v){0.f,0.f,0.f,0.f,0.f,0.f,0.f,0.f};

    for (int k0 = 0; k0 < Hh; k0 += 32) {
        // A fragments (16x32 f16): lanes 0-15 K=[0..7],[16..23]; lanes 16-31 K=[8..15],[24..31]
        union { h16v v; h8v h[2]; } A0, A1;
        const _Float16* ab0 = &hbuf[ml * HPAD + k0 + sel * 8];
        const _Float16* ab1 = &hbuf[(16 + ml) * HPAD + k0 + sel * 8];
        A0.h[0] = *(const h8v*)ab0;
        A0.h[1] = *(const h8v*)(ab0 + 16);
        A1.h[0] = *(const h8v*)ab1;
        A1.h[1] = *(const h8v*)(ab1 + 16);
        #pragma unroll
        for (int nt = 0; nt < 4; ++nt) {
            int n0 = w * 64 + nt * 16;
            // B fragment (32x16 f16): lane=n, lanes 0-15 K=[0..7],[16..23]; lanes 16-31 K=[8..15],[24..31]
            union { h16v v; h8v h[2]; } Bm;
            const _Float16* bb = pe2h + (size_t)(n0 + ml) * Hh + k0 + sel * 8;
            Bm.h[0] = *(const h8v*)bb;
            Bm.h[1] = *(const h8v*)(bb + 16);
            acc[0][nt] = __builtin_amdgcn_wmma_f32_16x16x32_f16(
                false, A0.v, false, Bm.v, (short)0, acc[0][nt], false, false);
            acc[1][nt] = __builtin_amdgcn_wmma_f32_16x16x32_f16(
                false, A1.v, false, Bm.v, (short)0, acc[1][nt], false, false);
        }
    }
    __syncthreads();   // all hbuf reads done -> safe to reuse for enc

    // write enc (+bias) into hbuf as f16 for the head matmul
    #pragma unroll
    for (int mt = 0; mt < 2; ++mt) {
        #pragma unroll
        for (int nt = 0; nt < 4; ++nt) {
            int nc = w * 64 + nt * 16 + ml;
            float bias = pe2_b[nc];
            #pragma unroll
            for (int rv = 0; rv < 8; ++rv) {
                int m = mt * 16 + (sel ? (8 + rv) : rv);   // D layout: VGPR rv -> M=rv / M=8+rv
                hbuf[m * HPAD + nc] = (_Float16)(acc[mt][nt][rv] + bias);
            }
        }
    }
    __syncthreads();

    // ---- phase 3: logits = enc @ [pred_w; ex_w]^T  (waves 0-3: 2 Mtiles x 2 Ntiles) ----
    if (w < 4) {
        int mt = w >> 1;           // M-tile 0/1
        int n0 = (w & 1) * 16;     // output cols 0-15 / 16-31
        f8v acc2 = (f8v){0.f,0.f,0.f,0.f,0.f,0.f,0.f,0.f};
        for (int k0 = 0; k0 < Hh; k0 += 32) {
            union { h16v v; h8v h[2]; } A;
            const _Float16* ab = &hbuf[(mt * 16 + ml) * HPAD + k0 + sel * 8];
            A.h[0] = *(const h8v*)ab;
            A.h[1] = *(const h8v*)(ab + 16);
            union { h16v v; h8v h[2]; } Bm;
            const _Float16* bb = pwh + (size_t)(n0 + ml) * Hh + k0 + sel * 8;
            Bm.h[0] = *(const h8v*)bb;
            Bm.h[1] = *(const h8v*)(bb + 16);
            acc2 = __builtin_amdgcn_wmma_f32_16x16x32_f16(
                false, A.v, false, Bm.v, (short)0, acc2, false, false);
        }
        int col = n0 + ml;  // 0..31
        #pragma unroll
        for (int rv = 0; rv < 8; ++rv) {
            int m  = mt * 16 + (sel ? (8 + rv) : rv);
            int rr = r0 + m;
            if (rr < R) {
                float v = acc2[rv];
                if (col < 26)       out_pred[(size_t)rr * 26 + col] = v + pred_b[col];
                else if (col == 26) out_ex[rr] = v + ex_b[0];
            }
        }
    }
}

// =======================================================================
extern "C" void kernel_launch(void* const* d_in, const int* in_sizes, int n_in,
                              void* d_out, int out_size, void* d_ws, size_t ws_size,
                              hipStream_t stream) {
    const float* temporal = (const float*)d_in[0];   // (16,10,512)
    const float* objf     = (const float*)d_in[1];   // (16,20,512)
    const float* fq       = (const float*)d_in[2];   // (5,512)
    const float* sa_in_w  = (const float*)d_in[3];
    const float* sa_in_b  = (const float*)d_in[4];
    const float* sa_out_w = (const float*)d_in[5];
    const float* sa_out_b = (const float*)d_in[6];
    const float* ca_in_w  = (const float*)d_in[7];
    const float* ca_in_b  = (const float*)d_in[8];
    const float* ca_out_w = (const float*)d_in[9];
    const float* ca_out_b = (const float*)d_in[10];
    const float* ff1_w    = (const float*)d_in[11];
    const float* ff1_b    = (const float*)d_in[12];
    const float* ff2_w    = (const float*)d_in[13];
    const float* ff2_b    = (const float*)d_in[14];
    const float* n1_w     = (const float*)d_in[15];
    const float* n1_b     = (const float*)d_in[16];
    const float* n2_w     = (const float*)d_in[17];
    const float* n2_b     = (const float*)d_in[18];
    const float* n3_w     = (const float*)d_in[19];
    const float* n3_b     = (const float*)d_in[20];
    const float* norm_w   = (const float*)d_in[21];
    const float* norm_b   = (const float*)d_in[22];
    const float* pe1_w    = (const float*)d_in[23];  // (512,1536)
    const float* pe1_b    = (const float*)d_in[24];
    const float* pe2_w    = (const float*)d_in[25];  // (512,512)
    const float* pe2_b    = (const float*)d_in[26];
    const float* pred_w   = (const float*)d_in[27];  // (26,512)
    const float* pred_b   = (const float*)d_in[28];
    const float* ex_w     = (const float*)d_in[29];  // (1,512)
    const float* ex_b     = (const float*)d_in[30];
    (void)in_sizes; (void)n_in; (void)ws_size;

    // Derive Fh from output size (graph-capture safe; no device readback).
    int F = out_size / (Bb * Pp * 27);
    if (F < 1) F = 1;
    if (F > 5) F = 5;
    const int BF = Bb * F;
    const int R  = Bb * F * Pp;

    // workspace bump allocator
    char* w8 = (char*)d_ws;
    size_t off = 0;
    auto alloc = [&](size_t bytes) -> void* {
        void* p = w8 + off;
        off += (bytes + 255) & ~(size_t)255;
        return p;
    };
    float* x        = (float*)alloc((size_t)BF * Hh * 4);
    float* qkv_x    = (float*)alloc((size_t)BF * 3 * Hh * 4);
    float* qkv_mem  = (float*)alloc((size_t)Bb * 10 * 3 * Hh * 4);
    float* attn_out = (float*)alloc((size_t)BF * Hh * 4);
    float* proj     = (float*)alloc((size_t)BF * Hh * 4);
    float* h1       = (float*)alloc((size_t)BF * FFh * 4);
    float* ctx      = (float*)alloc((size_t)BF * Hh * 4);
    float* Sbuf     = (float*)alloc((size_t)Bb * NOBJ * Hh * 4);
    float* Obuf     = (float*)alloc((size_t)Bb * NOBJ * Hh * 4);
    float* Cbuf     = (float*)alloc((size_t)BF * Hh * 4);
    _Float16* pe2h  = (_Float16*)alloc((size_t)Hh * Hh * 2);
    _Float16* pwh   = (_Float16*)alloc((size_t)32 * Hh * 2);

    auto cdiv = [](int a, int b) { return (a + b - 1) / b; };

    // ---- init x ----
    init_x_kernel<<<cdiv(BF * Hh, 256), 256, 0, stream>>>(fq, x, F);

    // ---- decoder layers ----
    for (int l = 0; l < 2; ++l) {
        // self-attention
        gemm_bias_act<<<cdiv(BF * 3 * Hh, 256), 256, 0, stream>>>(
            x, Hh, sa_in_w + (size_t)l * 3 * Hh * Hh, Hh, 0, sa_in_b + (size_t)l * 3 * Hh,
            qkv_x, 3 * Hh, BF, 3 * Hh, Hh, 0);
        attn_kernel<<<cdiv(Bb * NHh * F, 128), 128, 0, stream>>>(
            qkv_x, 3 * Hh, 0, qkv_x, 3 * Hh, Hh, 2 * Hh, attn_out, F, F);
        gemm_bias_act<<<cdiv(BF * Hh, 256), 256, 0, stream>>>(
            attn_out, Hh, sa_out_w + (size_t)l * Hh * Hh, Hh, 0, sa_out_b + (size_t)l * Hh,
            proj, Hh, BF, Hh, Hh, 0);
        ln_kernel<<<BF, 256, 0, stream>>>(x, proj, n1_w + l * Hh, n1_b + l * Hh, x, BF);

        // cross-attention
        gemm_bias_act<<<cdiv(BF * 3 * Hh, 256), 256, 0, stream>>>(
            x, Hh, ca_in_w + (size_t)l * 3 * Hh * Hh, Hh, 0, ca_in_b + (size_t)l * 3 * Hh,
            qkv_x, 3 * Hh, BF, 3 * Hh, Hh, 0);
        gemm_bias_act<<<cdiv(Bb * 10 * 3 * Hh, 256), 256, 0, stream>>>(
            temporal, Hh, ca_in_w + (size_t)l * 3 * Hh * Hh, Hh, 0, ca_in_b + (size_t)l * 3 * Hh,
            qkv_mem, 3 * Hh, Bb * 10, 3 * Hh, Hh, 0);
        attn_kernel<<<cdiv(Bb * NHh * F, 128), 128, 0, stream>>>(
            qkv_x, 3 * Hh, 0, qkv_mem, 3 * Hh, Hh, 2 * Hh, attn_out, F, 10);
        gemm_bias_act<<<cdiv(BF * Hh, 256), 256, 0, stream>>>(
            attn_out, Hh, ca_out_w + (size_t)l * Hh * Hh, Hh, 0, ca_out_b + (size_t)l * Hh,
            proj, Hh, BF, Hh, Hh, 0);
        ln_kernel<<<BF, 256, 0, stream>>>(x, proj, n2_w + l * Hh, n2_b + l * Hh, x, BF);

        // FFN
        gemm_bias_act<<<cdiv(BF * FFh, 256), 256, 0, stream>>>(
            x, Hh, ff1_w + (size_t)l * FFh * Hh, Hh, 0, ff1_b + (size_t)l * FFh,
            h1, FFh, BF, FFh, Hh, 1);
        gemm_bias_act<<<cdiv(BF * Hh, 256), 256, 0, stream>>>(
            h1, FFh, ff2_w + (size_t)l * Hh * FFh, FFh, 0, ff2_b + (size_t)l * Hh,
            proj, Hh, BF, Hh, FFh, 0);
        ln_kernel<<<BF, 256, 0, stream>>>(x, proj, n3_w + l * Hh, n3_b + l * Hh, x, BF);
    }

    // ---- final norm -> ctx ----
    ln_kernel<<<BF, 256, 0, stream>>>(x, nullptr, norm_w, norm_b, ctx, BF);

    // ---- factored pe1 projections ----
    gemm_bias_act<<<cdiv(Bb * NOBJ * Hh, 256), 256, 0, stream>>>(
        objf, Hh, pe1_w, 3 * Hh, 0, nullptr, Sbuf, Hh, Bb * NOBJ, Hh, Hh, 0);
    gemm_bias_act<<<cdiv(Bb * NOBJ * Hh, 256), 256, 0, stream>>>(
        objf, Hh, pe1_w, 3 * Hh, Hh, nullptr, Obuf, Hh, Bb * NOBJ, Hh, Hh, 0);
    gemm_bias_act<<<cdiv(BF * Hh, 256), 256, 0, stream>>>(
        ctx, Hh, pe1_w, 3 * Hh, 2 * Hh, pe1_b, Cbuf, Hh, BF, Hh, Hh, 0);  // pe1_b folded here

    // ---- f16 weight prep ----
    cvt_f16_kernel<<<cdiv(Hh * Hh, 256), 256, 0, stream>>>(pe2_w, pe2h, Hh * Hh);
    build_head_kernel<<<cdiv(32 * Hh, 256), 256, 0, stream>>>(pred_w, ex_w, pwh);

    // ---- fused pair + heads (WMMA) ----
    float* out_pred = (float*)d_out;
    float* out_ex   = out_pred + (size_t)R * 26;
    pair_head_kernel<<<cdiv(R, TM), 256, 0, stream>>>(
        Sbuf, Obuf, Cbuf, pe2_b, pe2h, pwh, pred_b, ex_b, out_pred, out_ex, F, R);
}
